// PointCloudDecoder_46471546143490
// MI455X (gfx1250) — matrix operand order repaired and checked by reading, not compiled, not run on hardware
//
#include <hip/hip_runtime.h>
#include <hip/hip_bf16.h>

typedef __attribute__((ext_vector_type(16))) _Float16 v16h;
typedef __attribute__((ext_vector_type(8)))  _Float16 v8h;
typedef __attribute__((ext_vector_type(8)))  float    v8f;

// ---------------------------------------------------------------------------
// ds_swizzle XOR-butterfly max within 16-lane halves (xor masks < 16 never
// cross the half boundary).  Group-of-32 mode: and=0x1F, or=0, xor=XMASK.
// ---------------------------------------------------------------------------
template <int XMASK>
__device__ __forceinline__ float max_swz(float x)
{
    int y = __builtin_amdgcn_ds_swizzle(__float_as_int(x), 0x1F | (XMASK << 10));
    return fmaxf(x, __int_as_float(y));
}

// ---------------------------------------------------------------------------
// Self-KNN, k=16.  grid = (N/128, B), block = 128.  N is a multiple of 128.
// ---------------------------------------------------------------------------
#define KNN_BS 128

__global__ __launch_bounds__(KNN_BS)
void knn16_self_kernel(const float* __restrict__ xyz, int N, int* __restrict__ idx_out)
{
    const int b  = blockIdx.y;
    const int qi = blockIdx.x * KNN_BS + threadIdx.x;
    const float* base = xyz + (size_t)b * N * 3;

    const float qx = base[qi * 3 + 0];
    const float qy = base[qi * 3 + 1];
    const float qz = base[qi * 3 + 2];

    float bd[16]; int bi[16];
#pragma unroll
    for (int r = 0; r < 16; ++r) { bd[r] = 3.0e38f; bi[r] = 0; }

    __shared__ float sx[KNN_BS], sy[KNN_BS], sz[KNN_BS];

    for (int t0 = 0; t0 < N; t0 += KNN_BS) {
        __syncthreads();
        const int j = t0 + threadIdx.x;
        sx[threadIdx.x] = base[j * 3 + 0];
        sy[threadIdx.x] = base[j * 3 + 1];
        sz[threadIdx.x] = base[j * 3 + 2];
        if (t0 + KNN_BS < N)
            __builtin_prefetch(base + (size_t)(t0 + KNN_BS) * 3, 0, 0);
        __syncthreads();

        for (int t = 0; t < KNN_BS; ++t) {
            const float dx = qx - sx[t];
            const float dy = qy - sy[t];
            const float dz = qz - sz[t];
            const float d2 = dx * dx + dy * dy + dz * dz;
            if (d2 < bd[15]) {
                float d = d2; int ii = t0 + t;
#pragma unroll
                for (int r = 0; r < 16; ++r) {
                    const bool sw = d < bd[r];
                    const float td = bd[r]; const int ti = bi[r];
                    bd[r] = sw ? d  : td;  bi[r] = sw ? ii : ti;
                    d     = sw ? td : d;   ii    = sw ? ti : ii;
                }
            }
        }
    }
#pragma unroll
    for (int r = 0; r < 16; ++r)
        idx_out[((size_t)b * N + qi) * 16 + r] = bi[r];
}

// ---------------------------------------------------------------------------
// KNN k=3 (query != support) + normalized inverse-distance weights.
// ---------------------------------------------------------------------------
__global__ __launch_bounds__(KNN_BS)
void knn3_weights_kernel(const float* __restrict__ q_xyz, const float* __restrict__ s_xyz,
                         int Nq, int Ns, int* __restrict__ idx_out, float* __restrict__ w_out)
{
    const int b  = blockIdx.y;
    const int qi = blockIdx.x * KNN_BS + threadIdx.x;
    const float* qb = q_xyz + (size_t)b * Nq * 3;
    const float* sb = s_xyz + (size_t)b * Ns * 3;

    const float qx = qb[qi * 3 + 0];
    const float qy = qb[qi * 3 + 1];
    const float qz = qb[qi * 3 + 2];

    float bd[3]; int bi[3];
#pragma unroll
    for (int r = 0; r < 3; ++r) { bd[r] = 3.0e38f; bi[r] = 0; }

    __shared__ float sx[KNN_BS], sy[KNN_BS], sz[KNN_BS];

    for (int t0 = 0; t0 < Ns; t0 += KNN_BS) {
        __syncthreads();
        const int j = t0 + threadIdx.x;
        sx[threadIdx.x] = sb[j * 3 + 0];
        sy[threadIdx.x] = sb[j * 3 + 1];
        sz[threadIdx.x] = sb[j * 3 + 2];
        __syncthreads();

        for (int t = 0; t < KNN_BS; ++t) {
            const float dx = qx - sx[t];
            const float dy = qy - sy[t];
            const float dz = qz - sz[t];
            float d = dx * dx + dy * dy + dz * dz;
            int ii = t0 + t;
            if (d < bd[2]) {
#pragma unroll
                for (int r = 0; r < 3; ++r) {
                    const bool sw = d < bd[r];
                    const float td = bd[r]; const int ti = bi[r];
                    bd[r] = sw ? d  : td;  bi[r] = sw ? ii : ti;
                    d     = sw ? td : d;   ii    = sw ? ti : ii;
                }
            }
        }
    }

    float w0 = 1.0f / (bd[0] + 1e-8f);
    float w1 = 1.0f / (bd[1] + 1e-8f);
    float w2 = 1.0f / (bd[2] + 1e-8f);
    const float inv = 1.0f / (w0 + w1 + w2);
    const size_t o = ((size_t)b * Nq + qi) * 3;
    idx_out[o + 0] = bi[0]; idx_out[o + 1] = bi[1]; idx_out[o + 2] = bi[2];
    w_out[o + 0] = w0 * inv; w_out[o + 1] = w1 * inv; w_out[o + 2] = w2 * inv;
}

// ---------------------------------------------------------------------------
// Inverse-distance interpolation gather: f_in [B,C,Ns] -> f_out [B,C,Nq].
// ---------------------------------------------------------------------------
__global__ void interp_gather_kernel(const float* __restrict__ f_in,
                                     const int* __restrict__ idx3,
                                     const float* __restrict__ w3,
                                     int C, int Ns, int Nq,
                                     float* __restrict__ f_out, int total)
{
    const int i = blockIdx.x * blockDim.x + threadIdx.x;
    if (i >= total) return;
    const int q = i % Nq;
    const int c = (i / Nq) % C;
    const int b = i / (Nq * C);
    const size_t p = ((size_t)b * Nq + q) * 3;
    const int*   id = idx3 + p;
    const float* w  = w3 + p;
    const float* fb = f_in + ((size_t)b * C + c) * Ns;
    f_out[i] = w[0] * fb[id[0]] + w[1] * fb[id[1]] + w[2] * fb[id[2]];
}

// ---------------------------------------------------------------------------
// Weight pre-pack: f32 -> f16 (row-major layout is already WMMA-A friendly:
// each lane's A slots are two contiguous 8-half runs -> global_load_b128).
// ---------------------------------------------------------------------------
__global__ void pack_w_kernel(const float* __restrict__ W, _Float16* __restrict__ Wp, int total)
{
    const int i = blockIdx.x * blockDim.x + threadIdx.x;
    if (i < total) Wp[i] = (_Float16)W[i];
}

// ---------------------------------------------------------------------------
// Edge-conv via WMMA, packed operands.
//   E tile staged in LDS as [neighbor][K] (K contiguous, f_i rows baked in,
//   row stride padded +8 halves = +4 banks -> conflict-free b128 reads).
//   Inner loop per 32-wide K chunk: 2x global_load_b128 (A) +
//   2x ds_load_b128 (B) + 1x v_wmma_f32_16x16x32_f16.
// grid = (N, B), block = (C_OUT/16) waves.
// ---------------------------------------------------------------------------
template <int C_IN, int C_OUT, bool RES>
__global__ __launch_bounds__((C_OUT / 16) * 32)
void edgeconv_kernel(const float* __restrict__ f_in,      // [B, C_IN, N]
                     const int*   __restrict__ nbr,       // [B, N, 16]
                     const _Float16* __restrict__ Wp,     // [C_OUT, 2*C_IN] f16
                     const float* __restrict__ bias,      // [C_OUT]
                     float* __restrict__ f_out,           // [B, C_OUT, N]
                     int N)
{
    constexpr int KTOT = 2 * C_IN;
    constexpr int LDK  = KTOT + 8;     // padded row stride in halves
    const int m   = blockIdx.x;
    const int b   = blockIdx.y;
    const int tid = threadIdx.x;

    __shared__ __align__(16) _Float16 sh_e[16 * LDK];
    __shared__ float sh_fif[C_IN];
    __shared__ int   sh_idx[16];

    const float* fb = f_in + (size_t)b * C_IN * N;

    if (tid < 16) sh_idx[tid] = nbr[((size_t)b * N + m) * 16 + tid];
    for (int c = tid; c < C_IN; c += blockDim.x)
        sh_fif[c] = fb[(size_t)c * N + m];
    __syncthreads();
    for (int i = tid; i < 16 * KTOT; i += blockDim.x) {
        const int n = i / KTOT;
        const int c = i & (KTOT - 1);
        const float v = (c < C_IN) ? (fb[(size_t)c * N + sh_idx[n]] - sh_fif[c])
                                   : sh_fif[c - C_IN];
        sh_e[n * LDK + c] = (_Float16)v;
    }
    __syncthreads();

    const int wave = tid >> 5;
    const int lane = tid & 31;
    const int half = lane >> 4;        // 0: lanes 0-15, 1: lanes 16-31
    const int l16  = lane & 15;
    const int ob   = wave * 16;
    const int orow = ob + l16;         // A-matrix M index (row of W)

    const _Float16* wrow = Wp   + (size_t)orow * KTOT;
    const _Float16* erow = sh_e + l16 * LDK;

    v8f acc = {};
    for (int kb = 0; kb < KTOT; kb += 32) {
        union { v16h v; v8h h[2]; } a, e;
        // A slots 0..7 : K = kb + 8*half + [0..7]
        // A slots 8..15: K = kb + 16 + 8*half + [0..7]
        a.h[0] = *(const v8h*)(wrow + kb + 8 * half);
        a.h[1] = *(const v8h*)(wrow + kb + 16 + 8 * half);
        // B slots 0..7 : K = kb + 8*half + [0..7]; slots 8..15: K = kb + 16 + 8*half + [0..7], N = l16
        e.h[0] = *(const v8h*)(erow + kb + 8 * half);
        e.h[1] = *(const v8h*)(erow + kb + 16 + 8 * half);
        acc = __builtin_amdgcn_wmma_f32_16x16x32_f16(
            /*neg_a=*/false, a.v, /*neg_b=*/false, e.v,
            /*c_mod=*/(short)0, acc, /*reuse_a=*/false, /*reuse_b=*/false);
    }

    // D tile: lane(l16)=neighbor n, VGPR v -> out channel ob + v + 8*half.
    // Max over 16 neighbors = ds_swizzle XOR butterfly inside each half.
#pragma unroll
    for (int v = 0; v < 8; ++v) {
        float mv = acc[v];
        mv = max_swz<1>(mv);
        mv = max_swz<2>(mv);
        mv = max_swz<4>(mv);
        mv = max_swz<8>(mv);
        if (l16 == 0) {
            const int o = ob + v + half * 8;
            float r = fmaxf(mv + bias[o], 0.0f);   // bias & relu commute with max
            if (RES) r += fb[(size_t)o * N + m];   // C_OUT == C_IN when RES
            f_out[((size_t)b * C_OUT + o) * N + m] = r;
        }
    }
}

// ---------------------------------------------------------------------------
// Final 3x64 projection.
// ---------------------------------------------------------------------------
__global__ __launch_bounds__(128)
void out_proj_kernel(const float* __restrict__ f, const float* __restrict__ Wo,
                     const float* __restrict__ bo, int N, float* __restrict__ out)
{
    const int b = blockIdx.y;
    const int n = blockIdx.x * blockDim.x + threadIdx.x;
    float a0 = bo[0], a1 = bo[1], a2 = bo[2];
    const float* fbase = f + (size_t)b * 64 * N;
#pragma unroll 4
    for (int c = 0; c < 64; ++c) {
        const float v = fbase[(size_t)c * N + n];
        a0 += Wo[c]       * v;
        a1 += Wo[64 + c]  * v;
        a2 += Wo[128 + c] * v;
    }
    out[((size_t)b * 3 + 0) * N + n] = a0;
    out[((size_t)b * 3 + 1) * N + n] = a1;
    out[((size_t)b * 3 + 2) * N + n] = a2;
}

// ---------------------------------------------------------------------------
extern "C" void kernel_launch(void* const* d_in, const int* in_sizes, int n_in,
                              void* d_out, int out_size, void* d_ws, size_t ws_size,
                              hipStream_t stream)
{
    (void)in_sizes; (void)n_in; (void)out_size; (void)ws_size;

    const float* xyz0   = (const float*)d_in[0];
    const float* xyz1   = (const float*)d_in[1];
    const float* xyz2   = (const float*)d_in[2];
    const float* feats  = (const float*)d_in[3];
    const float* W_l0b0 = (const float*)d_in[4];
    const float* b_l0b0 = (const float*)d_in[5];
    const float* W_l0up = (const float*)d_in[6];
    const float* b_l0up = (const float*)d_in[7];
    const float* W_l1b0 = (const float*)d_in[8];
    const float* b_l1b0 = (const float*)d_in[9];
    const float* W_l1up = (const float*)d_in[10];
    const float* b_l1up = (const float*)d_in[11];
    const float* W_out  = (const float*)d_in[12];
    const float* b_out  = (const float*)d_in[13];

    constexpr int B = 2, N0 = 16384, N1 = 4096, N2 = 1024;
    constexpr int C = 256, D1 = 128, D2 = 64;

    char* ws = (char*)d_ws;
    size_t off = 0;
    auto carve = [&](size_t bytes) -> void* {
        void* p = ws + off;
        off = (off + bytes + 255) & ~(size_t)255;
        return p;
    };

    int*   idx2    = (int*)  carve((size_t)B * N2 * 16 * sizeof(int));
    int*   idx1    = (int*)  carve((size_t)B * N1 * 16 * sizeof(int));
    int*   idx0    = (int*)  carve((size_t)B * N0 * 16 * sizeof(int));
    int*   i21_idx = (int*)  carve((size_t)B * N1 * 3 * sizeof(int));
    float* i21_w   = (float*)carve((size_t)B * N1 * 3 * sizeof(float));
    int*   i10_idx = (int*)  carve((size_t)B * N0 * 3 * sizeof(int));
    float* i10_w   = (float*)carve((size_t)B * N0 * 3 * sizeof(float));
    float* fA      = (float*)carve((size_t)B * C  * N2 * sizeof(float));
    float* fB      = (float*)carve((size_t)B * C  * N1 * sizeof(float));
    float* fC      = (float*)carve((size_t)B * D1 * N1 * sizeof(float));
    float* fD      = (float*)carve((size_t)B * D1 * N1 * sizeof(float));
    float* fE      = (float*)carve((size_t)B * D1 * N0 * sizeof(float));
    float* fF      = (float*)carve((size_t)B * D2 * N0 * sizeof(float));
    _Float16* Wp0  = (_Float16*)carve((size_t)C  * 2 * C  * sizeof(_Float16)); // 256 x 512
    _Float16* Wp1  = (_Float16*)carve((size_t)D1 * 2 * C  * sizeof(_Float16)); // 128 x 512
    _Float16* Wp2  = (_Float16*)carve((size_t)D1 * 2 * D1 * sizeof(_Float16)); // 128 x 256
    _Float16* Wp3  = (_Float16*)carve((size_t)D2 * 2 * D1 * sizeof(_Float16)); //  64 x 256

    // ---- weight pre-pack (f32 -> f16, WMMA-A friendly row-major) ----
    pack_w_kernel<<<(C * 2 * C + 255) / 256, 256, 0, stream>>>(W_l0b0, Wp0, C * 2 * C);
    pack_w_kernel<<<(D1 * 2 * C + 255) / 256, 256, 0, stream>>>(W_l0up, Wp1, D1 * 2 * C);
    pack_w_kernel<<<(D1 * 2 * D1 + 255) / 256, 256, 0, stream>>>(W_l1b0, Wp2, D1 * 2 * D1);
    pack_w_kernel<<<(D2 * 2 * D1 + 255) / 256, 256, 0, stream>>>(W_l1up, Wp3, D2 * 2 * D1);

    // ---- stage 1: edge conv on N2 (256 -> 256, residual) ----
    knn16_self_kernel<<<dim3(N2 / KNN_BS, B), KNN_BS, 0, stream>>>(xyz2, N2, idx2);
    edgeconv_kernel<C, C, true><<<dim3(N2, B), (C / 16) * 32, 0, stream>>>(
        feats, idx2, Wp0, b_l0b0, fA, N2);

    // ---- stage 2: interpolate N2 -> N1 ----
    knn3_weights_kernel<<<dim3(N1 / KNN_BS, B), KNN_BS, 0, stream>>>(
        xyz1, xyz2, N1, N2, i21_idx, i21_w);
    {
        const int total = B * C * N1;
        interp_gather_kernel<<<(total + 255) / 256, 256, 0, stream>>>(
            fA, i21_idx, i21_w, C, N2, N1, fB, total);
    }

    // ---- stage 3+4: edge convs on N1 (256 -> 128, then 128 -> 128 residual) ----
    knn16_self_kernel<<<dim3(N1 / KNN_BS, B), KNN_BS, 0, stream>>>(xyz1, N1, idx1);
    edgeconv_kernel<C, D1, false><<<dim3(N1, B), (D1 / 16) * 32, 0, stream>>>(
        fB, idx1, Wp1, b_l0up, fC, N1);
    edgeconv_kernel<D1, D1, true><<<dim3(N1, B), (D1 / 16) * 32, 0, stream>>>(
        fC, idx1, Wp2, b_l1b0, fD, N1);

    // ---- stage 5: interpolate N1 -> N0 ----
    knn3_weights_kernel<<<dim3(N0 / KNN_BS, B), KNN_BS, 0, stream>>>(
        xyz0, xyz1, N0, N1, i10_idx, i10_w);
    {
        const int total = B * D1 * N0;
        interp_gather_kernel<<<(total + 255) / 256, 256, 0, stream>>>(
            fD, i10_idx, i10_w, D1, N1, N0, fE, total);
    }

    // ---- stage 6: edge conv on N0 (128 -> 64) ----
    knn16_self_kernel<<<dim3(N0 / KNN_BS, B), KNN_BS, 0, stream>>>(xyz0, N0, idx0);
    edgeconv_kernel<D1, D2, false><<<dim3(N0, B), (D2 / 16) * 32, 0, stream>>>(
        fE, idx0, Wp3, b_l1up, fF, N0);

    // ---- stage 7: output projection ----
    out_proj_kernel<<<dim3(N0 / 128, B), 128, 0, stream>>>(
        fF, W_out, b_out, N0, (float*)d_out);
}
